// LG_VSSBlock_8392366097073
// MI455X (gfx1250) — compile-verified, matrix operand order checked by reading, not run on hardware
//
#include <hip/hip_runtime.h>
#include <hip/hip_bf16.h>

// ---------------------------------------------------------------------------
// LG_VSSBlock for MI455X (gfx1250, wave32, WMMA).
// All large GEMMs use v_wmma_f32_16x16x32_bf16 (bf16 A/B, f32 accum), with
// LDS tiles stored in *fragment order* so each operand is a single 32B
// ds_load_b128 pair per lane.
// Selective scan uses a 3-phase chunked parallel scan (linear diagonal rec.).
// ---------------------------------------------------------------------------

typedef __attribute__((ext_vector_type(16))) __bf16 bf16x16;
typedef __attribute__((ext_vector_type(2)))  __bf16 bf16x2;
typedef __attribute__((ext_vector_type(8)))  float  f32x8;

#define NTOK 16384   // B*H*W tokens
#define HWSZ 4096    // H*W
#define CCH  128     // C
#define DICH 256     // DI
#define LCH  64      // scan chunk length
#define NCH  64      // chunks per batch (4096/64)

// ------------------------------- generic WMMA GEMM -------------------------
// C[m,n] = sum_k A[m,k]*B[k,n]  (+bias[m]) with flexible B/C addressing.
// Block tile 64(M) x 64(N), k-step 32, 4 waves (128 threads).
struct GemmP {
    const float* A; int lda;                 // A row-major MxK
    const float* B; long ldbk; long ldbn;    // B element (k,n) = B[k*ldbk+n*ldbn]
    int bmode;                               // 0: strided  1: BCHW (C=128,HW=4096)
    float* C; long ldcm; long ldcn;
    const float* bias;                       // may be null (indexed by m)
    const float* resid;                      // emode 2 only (BCHW layout)
    int M; long N; int K;
    int emode;                               // 0: strided  1: BCHW+bias  2: BCHW+bias+resid
};

// fragment layout (wave32, 16-bit A/B): lane = (row%16) + 16*khi,
// element e = ((k>>4)<<3) + (k&7), khi = (k>>3)&1.  Stored pre-permuted.
__global__ __launch_bounds__(128) void gemm_wmma(GemmP p) {
    __shared__ bf16x16 Afrag[4][32];   // 4 m-tiles x 32 lanes (4 KB)
    __shared__ bf16x16 Bfrag[4][32];   // 4 n-tiles x 32 lanes (4 KB)
    const int tid  = threadIdx.x;
    const int lane = tid & 31;
    const int wave = tid >> 5;
    const long n0  = (long)blockIdx.x * 64;
    const int  m0  = blockIdx.y * 64;

    // staging roles
    const int am  = tid >> 1;     // A: row within tile (0..63)
    const int akh = tid & 1;      // A: k-half (16 each)
    const int bn  = tid & 63;     // B: column within tile (0..63)
    const int bkh = tid >> 6;     // B: k-half

    f32x8 acc[4] = {};

    for (int k0 = 0; k0 < p.K; k0 += 32) {
        // ---- stage A: 16 contiguous fp32 per thread, scatter bf16 pairs ----
        {
            float v[16];
            if (m0 + am < p.M) {
                const float4* s4 =
                    (const float4*)(p.A + (long)(m0 + am) * p.lda + k0 + akh * 16);
                #pragma unroll
                for (int q = 0; q < 4; ++q) {
                    float4 f = s4[q];
                    v[4*q+0] = f.x; v[4*q+1] = f.y; v[4*q+2] = f.z; v[4*q+3] = f.w;
                }
            } else {
                #pragma unroll
                for (int q = 0; q < 16; ++q) v[q] = 0.f;
            }
            const int tile = am >> 4, lmod = am & 15;
            #pragma unroll
            for (int jj = 0; jj < 16; jj += 2) {
                int kl = akh * 16 + jj;
                int ln = lmod + (((kl >> 3) & 1) << 4);
                int e  = ((kl >> 4) << 3) + (kl & 7);
                bf16x2 pr; pr[0] = (__bf16)v[jj]; pr[1] = (__bf16)v[jj+1];
                ((bf16x2*)&Afrag[tile][ln])[e >> 1] = pr;
            }
        }
        // ---- stage B: coalesced across the 64 columns, scatter bf16 pairs ----
        {
            const long n   = n0 + bn;
            const int tile = bn >> 4, lmodn = bn & 15;
            #pragma unroll
            for (int jj = 0; jj < 16; jj += 2) {
                int kl = bkh * 16 + jj;
                int k  = k0 + kl;
                long bi0, bi1;
                if (p.bmode == 1) {
                    long base = ((n >> 12) * CCH) << 12;
                    bi0 = base + ((long)k << 12) + (n & (HWSZ - 1));
                    bi1 = bi0 + HWSZ;                        // k+1
                } else {
                    bi0 = (long)k * p.ldbk + n * p.ldbn;
                    bi1 = bi0 + p.ldbk;
                }
                bf16x2 pr; pr[0] = (__bf16)p.B[bi0]; pr[1] = (__bf16)p.B[bi1];
                int ln = lmodn + (((kl >> 3) & 1) << 4);
                int e  = ((kl >> 4) << 3) + (kl & 7);
                ((bf16x2*)&Bfrag[tile][ln])[e >> 1] = pr;
            }
        }
        __syncthreads();

        // ---- fragments: one 32B vector read each (2 x ds_load_b128) ----
        bf16x16 a = Afrag[wave][lane];
        #pragma unroll
        for (int j = 0; j < 4; ++j)
            acc[j] = __builtin_amdgcn_wmma_f32_16x16x32_bf16(
                false, a, false, Bfrag[j][lane], (short)0, acc[j], false, false);
        __syncthreads();
    }

    // ---- epilogue: C layout = VGPR r -> M = r + 8*(lane>=16), N = lane&15 ----
    const int mrow = m0 + wave * 16 + ((lane >> 4) << 3);
    const int ncol = lane & 15;
    #pragma unroll
    for (int j = 0; j < 4; ++j) {
        long n = n0 + j * 16 + ncol;
        #pragma unroll
        for (int r = 0; r < 8; ++r) {
            int m = mrow + r;
            if (m >= p.M) continue;
            float v = acc[j][r];
            if (p.bias) v += p.bias[m];
            if (p.emode == 0) {
                p.C[(long)m * p.ldcm + n * p.ldcn] = v;
            } else {
                long oi = (((n >> 12) * CCH + m) << 12) + (n & (HWSZ - 1));
                if (p.emode == 2) v += p.resid[oi];
                p.C[oi] = v;
            }
        }
    }
}

// --------------------- local branch: dw3x3 + BN(eval) + SiLU ----------------
__global__ __launch_bounds__(256) void local_kernel(
    const float* __restrict__ xh, const float* __restrict__ w,
    const float* __restrict__ bng, const float* __restrict__ bnb,
    const float* __restrict__ bnm, const float* __restrict__ bnv,
    float* __restrict__ lg) {
    long id = (long)blockIdx.x * 256 + threadIdx.x;   // over 128*16384
    int c = (int)(id >> 14);
    int n = (int)(id & (NTOK - 1));
    int b = n >> 12, hw = n & (HWSZ - 1);
    int y = hw >> 6, x = hw & 63;
    const float* base = xh + (((long)b * CCH + c) << 12);
    float s = 0.f;
    #pragma unroll
    for (int ky = 0; ky < 3; ++ky) {
        int yy = y + ky - 1;
        if (yy < 0 || yy > 63) continue;
        #pragma unroll
        for (int kx = 0; kx < 3; ++kx) {
            int xx = x + kx - 1;
            if (xx < 0 || xx > 63) continue;
            s += w[c * 9 + ky * 3 + kx] * base[yy * 64 + xx];
        }
    }
    float sc = bng[c] * rsqrtf(bnv[c] + 1e-5f);
    float v = (s - bnm[c]) * sc + bnb[c];
    v = v / (1.f + __expf(-v));                       // SiLU
    lg[(long)c * NTOK + n] = v;
}

// --------------------------- token LayerNorm (C=128) ------------------------
__global__ __launch_bounds__(128) void ln_kernel(
    const float* __restrict__ xh, const float* __restrict__ g,
    const float* __restrict__ be, float* __restrict__ t) {
    int wave = threadIdx.x >> 5, lane = threadIdx.x & 31;
    long n = (long)blockIdx.x * 4 + wave;
    int b = (int)(n >> 12), hw = (int)(n & (HWSZ - 1));
    const float* base = xh + (((long)b * CCH) << 12) + hw;
    float v[4], s = 0.f, sq = 0.f;
    #pragma unroll
    for (int j = 0; j < 4; ++j) {
        float x = base[((long)(lane + 32 * j)) << 12];
        v[j] = x; s += x; sq += x * x;
    }
    #pragma unroll
    for (int m = 16; m >= 1; m >>= 1) {
        s  += __shfl_xor(s,  m, 32);
        sq += __shfl_xor(sq, m, 32);
    }
    float mu  = s * (1.f / 128.f);
    float var = sq * (1.f / 128.f) - mu * mu;
    float inv = rsqrtf(var + 1e-5f);
    #pragma unroll
    for (int j = 0; j < 4; ++j) {
        int c = lane + 32 * j;
        t[n * CCH + c] = (v[j] - mu) * inv * g[c] + be[c];
    }
}

// --------------------- causal depthwise conv1d (k=4) + SiLU -----------------
__global__ __launch_bounds__(256) void conv1d_kernel(
    const float* __restrict__ xz, const float* __restrict__ w,
    const float* __restrict__ cb, float* __restrict__ xmc) {
    long id = (long)blockIdx.x * 256 + threadIdx.x;   // over 256*16384
    int d = (int)(id >> 14);
    int n = (int)(id & (NTOK - 1));
    int lp = n & (HWSZ - 1);
    float s = cb[d];
    const float* row = xz + (long)d * NTOK;
    #pragma unroll
    for (int j = 0; j < 4; ++j) {
        int tp = lp - 3 + j;
        if (tp >= 0) s += w[d * 4 + j] * row[n - 3 + j];
    }
    s = s / (1.f + __expf(-s));                       // SiLU
    xmc[id] = s;
}

// ------------------------ dt_proj (K=8) + softplus --------------------------
__global__ __launch_bounds__(256) void dt_kernel(
    const float* __restrict__ dbl, const float* __restrict__ dtw,
    const float* __restrict__ dtb, float* __restrict__ dt) {
    long id = (long)blockIdx.x * 256 + threadIdx.x;   // over 256*16384
    int d = (int)(id >> 14);
    int n = (int)(id & (NTOK - 1));
    float a = dtb[d];
    #pragma unroll
    for (int r = 0; r < 8; ++r) a += dtw[d * 8 + r] * dbl[(long)r * NTOK + n];
    dt[id] = (a > 20.f) ? a : log1pf(__expf(a));      // softplus
}

// ----------------- scan phase 1: per-chunk (P = prod dA, s) -----------------
// block = 128 = 8 d-channels x 16 states; grid = (B*NCH, DI/8)
__global__ __launch_bounds__(128) void scan1(
    const float* __restrict__ dt, const float* __restrict__ xmc,
    const float* __restrict__ dbl, const float* __restrict__ A_log,
    float* __restrict__ P, float* __restrict__ S) {
    int s = threadIdx.x & 15;
    int d = blockIdx.y * 8 + (threadIdx.x >> 4);
    long n0 = (long)blockIdx.x * LCH;
    float A = -__expf(A_log[d * 16 + s]);
    const float* dtr = dt  + (long)d * NTOK;
    const float* ur  = xmc + (long)d * NTOK;
    const float* Br  = dbl + (long)(8 + s) * NTOK;
    float h = 0.f, Pp = 1.f;
    for (int t = 0; t < LCH; ++t) {
        long n = n0 + t;
        float dtv = dtr[n];
        float da  = __expf(dtv * A);
        h  = da * h + dtv * Br[n] * ur[n];
        Pp *= da;
    }
    long idx = (long)blockIdx.x * 4096 + d * 16 + s;  // chunk-major
    P[idx] = Pp; S[idx] = h;
}

// -------- scan phase 2: serial scan over 64 chunk summaries (parallel states)
__global__ __launch_bounds__(256) void scan2(
    const float* __restrict__ P, const float* __restrict__ S,
    float* __restrict__ H0) {
    int id = blockIdx.x * 256 + threadIdx.x;          // 16384 = b*4096 + d*16 + s
    int b = id >> 12, ds = id & 4095;
    float h = 0.f;
    for (int ch = 0; ch < NCH; ++ch) {
        long idx = ((long)(b * NCH + ch)) * 4096 + ds;
        H0[idx] = h;                                  // incoming state of chunk
        h = P[idx] * h + S[idx];
    }
}

// ------- scan phase 3: chunk rerun with true h0, y = C.h, gate, write yg ----
__global__ __launch_bounds__(128) void scan3(
    const float* __restrict__ dt, const float* __restrict__ xmc,
    const float* __restrict__ dbl, const float* __restrict__ A_log,
    const float* __restrict__ Dv, const float* __restrict__ H0,
    const float* __restrict__ xz, float* __restrict__ yg) {
    int s = threadIdx.x & 15;
    int d = blockIdx.y * 8 + (threadIdx.x >> 4);
    long n0 = (long)blockIdx.x * LCH;
    long idx = (long)blockIdx.x * 4096 + d * 16 + s;
    float A = -__expf(A_log[d * 16 + s]);
    float h = H0[idx];
    const float* dtr = dt  + (long)d * NTOK;
    const float* ur  = xmc + (long)d * NTOK;
    const float* Br  = dbl + (long)(8 + s)  * NTOK;
    const float* Cr  = dbl + (long)(24 + s) * NTOK;
    const float* zr  = xz  + (long)(DICH + d) * NTOK;
    float Dd = Dv[d];
    for (int t = 0; t < LCH; ++t) {
        long n = n0 + t;
        float dtv = dtr[n], u = ur[n];
        float da  = __expf(dtv * A);
        h = da * h + dtv * Br[n] * u;
        float part = h * Cr[n];
        part += __shfl_xor(part, 1, 32);              // reduce over DS=16 states
        part += __shfl_xor(part, 2, 32);
        part += __shfl_xor(part, 4, 32);
        part += __shfl_xor(part, 8, 32);
        if (s == 0) {
            float y = part + u * Dd;
            float z = zr[n];
            float g = z / (1.f + __expf(-z));         // silu(z)
            yg[(long)d * NTOK + n] = y * g;
        }
    }
}

// ---------------------------------------------------------------------------
extern "C" void kernel_launch(void* const* d_in, const int* in_sizes, int n_in,
                              void* d_out, int out_size, void* d_ws, size_t ws_size,
                              hipStream_t stream) {
    const float* x      = (const float*)d_in[0];
    const float* cv1_w  = (const float*)d_in[1];
    const float* cv1_b  = (const float*)d_in[2];
    const float* dw_w   = (const float*)d_in[3];
    const float* bn_g   = (const float*)d_in[4];
    const float* bn_b   = (const float*)d_in[5];
    const float* bn_m   = (const float*)d_in[6];
    const float* bn_v   = (const float*)d_in[7];
    const float* ln_g   = (const float*)d_in[8];
    const float* ln_b   = (const float*)d_in[9];
    const float* ipw    = (const float*)d_in[10];
    const float* c1w    = (const float*)d_in[11];
    const float* c1b    = (const float*)d_in[12];
    const float* xpw    = (const float*)d_in[13];
    const float* dtw    = (const float*)d_in[14];
    const float* dtbi   = (const float*)d_in[15];
    const float* A_log  = (const float*)d_in[16];
    const float* Dv     = (const float*)d_in[17];
    const float* opw    = (const float*)d_in[18];
    const float* cv2_w  = (const float*)d_in[19];
    const float* cv2_b  = (const float*)d_in[20];
    float* out = (float*)d_out;

    // workspace layout (floats); aliases exploit producer/consumer ordering
    float* ws  = (float*)d_ws;
    float* xh  = ws;                 // [128][B*HW] BCHW, 2,097,152
    float* tb  = xh  + 2097152;      // LN out, token-major [N][128]
    float* xz  = tb  + 2097152;      // [512][N] (xm rows 0-255, z rows 256-511)
    float* xmc = xz  + 8388608;      // [256][N] conv1d+silu
    float* dbp = xmc + 4194304;      // [40][N]  x_proj out (dt_raw|B|C)
    float* dtt = dbp + 655360;       // [256][N] dt after softplus
    float* lg  = dtt + 4194304;      // [256][N] concat(l, g) for cv2
    float* Pb  = xh;                 // alias: xh dead after local+LN
    float* Sb  = xh + 1048576;
    float* H0  = tb;                 // alias: t dead after in_proj GEMM
    float* yg  = xz;                 // alias: xm rows dead after conv1d

    // 1. cv1 1x1 conv: xh = cv1_w @ x  (BCHW in/out)
    GemmP p1{cv1_w, 128, x, 0, 0, 1, xh, 0, 0, cv1_b, nullptr, 128, NTOK, 128, 1};
    hipLaunchKernelGGL(gemm_wmma, dim3(256, 2), dim3(128), 0, stream, p1);

    // 2. local branch -> lg rows [0,128)
    hipLaunchKernelGGL(local_kernel, dim3(8192), dim3(256), 0, stream,
                       xh, dw_w, bn_g, bn_b, bn_m, bn_v, lg);

    // 3. LayerNorm -> t (token-major)
    hipLaunchKernelGGL(ln_kernel, dim3(4096), dim3(128), 0, stream,
                       xh, ln_g, ln_b, tb);

    // 4. in_proj: xz[j][n] = ipw[j][:] . t[n][:]
    GemmP p2{ipw, 128, tb, 1, 128, 0, xz, NTOK, 1, nullptr, nullptr, 512, NTOK, 128, 0};
    hipLaunchKernelGGL(gemm_wmma, dim3(256, 8), dim3(128), 0, stream, p2);

    // 5. causal depthwise conv1d + SiLU -> xmc
    hipLaunchKernelGGL(conv1d_kernel, dim3(16384), dim3(256), 0, stream,
                       xz, c1w, c1b, xmc);

    // 6. x_proj: dbl[j][n] = xpw[j][:] . xmc[:][n]   (M=40)
    GemmP p3{xpw, 256, xmc, NTOK, 1, 0, dbp, NTOK, 1, nullptr, nullptr, 40, NTOK, 256, 0};
    hipLaunchKernelGGL(gemm_wmma, dim3(256, 1), dim3(128), 0, stream, p3);

    // 7. dt_proj + softplus
    hipLaunchKernelGGL(dt_kernel, dim3(16384), dim3(256), 0, stream,
                       dbp, dtw, dtbi, dtt);

    // 8-10. chunked parallel selective scan
    hipLaunchKernelGGL(scan1, dim3(256, 32), dim3(128), 0, stream,
                       dtt, xmc, dbp, A_log, Pb, Sb);
    hipLaunchKernelGGL(scan2, dim3(64), dim3(256), 0, stream, Pb, Sb, H0);
    hipLaunchKernelGGL(scan3, dim3(256, 32), dim3(128), 0, stream,
                       dtt, xmc, dbp, A_log, Dv, H0, xz, yg);

    // 11. out_proj: g rows of lg [128,256) = opw @ yg
    GemmP p4{opw, 256, yg, NTOK, 1, 0, lg + (long)128 * NTOK, NTOK, 1,
             nullptr, nullptr, 128, NTOK, 256, 0};
    hipLaunchKernelGGL(gemm_wmma, dim3(256, 2), dim3(128), 0, stream, p4);

    // 12. cv2 merge + bias + residual -> d_out (BCHW)
    GemmP p5{cv2_w, 256, lg, NTOK, 1, 0, out, 0, 0, cv2_b, x, 128, NTOK, 256, 2};
    hipLaunchKernelGGL(gemm_wmma, dim3(256, 2), dim3(128), 0, stream, p5);

    (void)in_sizes; (void)n_in; (void)out_size; (void)ws_size;
}
